// GatedAttentionUnit_51436528337427
// MI455X (gfx1250) — matrix operand order repaired and checked by reading, not compiled, not run on hardware
//
#include <hip/hip_runtime.h>

// ---------------------------------------------------------------------------
// GAU forward on gfx1250 using v_wmma_f32_16x16x32_f16.
// Pipeline: f32->f16 converts; GEMM1(proj+silu+split); QK(+bias,mask,relu^2);
// v transpose; attn@V (+*u); output GEMM (+out_b).
// ---------------------------------------------------------------------------

typedef __attribute__((ext_vector_type(16))) _Float16 v16h;
typedef __attribute__((ext_vector_type(8)))  _Float16 v8h;
typedef __attribute__((ext_vector_type(8)))  float    v8f;
typedef __attribute__((ext_vector_type(4)))  float    v4f;

#define S_LEN 2048
#define BSZ   4
#define EDIM  1024
#define ZDIM  128
#define HDIM  2048
#define M1    (S_LEN * BSZ)        // 8192
#define N1    (2 * HDIM + ZDIM)    // 4224

__device__ __forceinline__ int imin(int a, int b) { return a < b ? a : b; }

// silu via hardware reciprocal: x * rcp(1 + exp(-x))  (single v_rcp_f32)
__device__ __forceinline__ float fast_silu(float t) {
  return t * __builtin_amdgcn_rcpf(1.0f + __expf(-t));
}

// ---- fragment loads (row-major, K contiguous) -----------------------------
__device__ __forceinline__ v16h ldfrag_a(const _Float16* __restrict__ A, size_t lda,
                                         int m0, int k0, int l16, int lh) {
  const _Float16* p = A + (size_t)(m0 + l16) * lda + (size_t)k0 + lh * 16;
  v8h lo = *reinterpret_cast<const v8h*>(p);
  v8h hi = *reinterpret_cast<const v8h*>(p + 8);
  return __builtin_shufflevector(lo, hi, 0,1,2,3,4,5,6,7,8,9,10,11,12,13,14,15);
}

__device__ __forceinline__ v16h ldfrag_b(const _Float16* __restrict__ BT, size_t ldb,
                                         int n0, int k0, int l16, int lh) {
  const _Float16* p = BT + (size_t)(n0 + l16) * ldb + (size_t)k0 + lh * 16;
  return *reinterpret_cast<const v16h*>(p);
}

// ---- wave-level 32x64 MMA inner loop --------------------------------------
template <int MF, int NF>
__device__ __forceinline__ void mma_loop(const _Float16* __restrict__ A, size_t lda,
                                         const _Float16* __restrict__ BT, size_t ldb,
                                         int m0, int n0, int kbeg, int kend,
                                         int l16, int lh, v8f (&acc)[MF][NF]) {
  for (int k0 = kbeg; k0 < kend; k0 += 32) {
    // speculative prefetch of next K-slab (non-faulting; no branch)
    __builtin_prefetch(A + (size_t)(m0 + l16) * lda + k0 + 32, 0, 2);
    __builtin_prefetch(BT + (size_t)(n0 + l16) * ldb + k0 + 32, 0, 2);
    v16h a[MF], b[NF];
#pragma unroll
    for (int i = 0; i < MF; ++i) a[i] = ldfrag_a(A, lda, m0 + 16 * i, k0, l16, lh);
#pragma unroll
    for (int j = 0; j < NF; ++j) b[j] = ldfrag_b(BT, ldb, n0 + 16 * j, k0, l16, lh);
#pragma unroll
    for (int i = 0; i < MF; ++i)
#pragma unroll
      for (int j = 0; j < NF; ++j)
        acc[i][j] = __builtin_amdgcn_wmma_f32_16x16x32_f16(
            false, a[i], false, b[j], (short)0, acc[i][j], false, false);
  }
}

// ---- f32 -> f16 convert (vectorized) --------------------------------------
__global__ __launch_bounds__(256) void f32_to_f16(const float* __restrict__ in,
                                                  _Float16* __restrict__ out, int n4) {
  for (int i = blockIdx.x * blockDim.x + threadIdx.x; i < n4; i += gridDim.x * blockDim.x) {
    v4f v = *reinterpret_cast<const v4f*>(in + (size_t)i * 4);
    _Float16* o = out + (size_t)i * 4;
    o[0] = (_Float16)v.x; o[1] = (_Float16)v.y;
    o[2] = (_Float16)v.z; o[3] = (_Float16)v.w;
  }
}

// ---- GEMM1: base = silu(x @ proj_w^T + b); split into u16/v16/q16/k16 ------
__global__ __launch_bounds__(256) void gemm1_kernel(
    const _Float16* __restrict__ A, const _Float16* __restrict__ BT,
    const float* __restrict__ bias, const float* __restrict__ gamma,
    const float* __restrict__ beta, _Float16* __restrict__ u16,
    _Float16* __restrict__ v16, _Float16* __restrict__ q16,
    _Float16* __restrict__ k16) {
  int tid = threadIdx.x, wave = tid >> 5, lane = tid & 31;
  int l16 = lane & 15, lh = lane >> 4;
  int m0 = blockIdx.y * 128 + (wave & 3) * 32;
  int n0 = blockIdx.x * 128 + (wave >> 2) * 64;
  v8f acc[2][4];
#pragma unroll
  for (int i = 0; i < 2; ++i)
#pragma unroll
    for (int j = 0; j < 4; ++j) acc[i][j] = (v8f){0,0,0,0,0,0,0,0};

  mma_loop<2, 4>(A, EDIM, BT, EDIM, m0, n0, 0, EDIM, l16, lh, acc);

#pragma unroll
  for (int i = 0; i < 2; ++i)
#pragma unroll
    for (int j = 0; j < 4; ++j) {
      int n = n0 + 16 * j + l16;
#pragma unroll
      for (int r = 0; r < 8; ++r) {
        int m = m0 + 16 * i + lh * 8 + r;
        float sv = fast_silu(acc[i][j][r] + bias[n]);
        int s = m >> 2, bb = m & 3;               // m = s*BSZ + b
        if (n < HDIM) {
          u16[(size_t)m * HDIM + n] = (_Float16)sv;
        } else if (n < 2 * HDIM) {
          v16[((size_t)bb * S_LEN + s) * HDIM + (n - HDIM)] = (_Float16)sv;
        } else {
          int zi = n - 2 * HDIM;
          size_t qi = ((size_t)bb * S_LEN + s) * ZDIM + zi;
          q16[qi] = (_Float16)(sv * gamma[zi] + beta[zi]);
          k16[qi] = (_Float16)(sv * gamma[ZDIM + zi] + beta[ZDIM + zi]);
        }
      }
    }
}

// ---- QK: attn = relu(qk/S + bias)^2 with causal mask, f16 ------------------
__global__ __launch_bounds__(256) void qk_kernel(const _Float16* __restrict__ q16,
                                                 const _Float16* __restrict__ k16,
                                                 const float* __restrict__ rpb,
                                                 _Float16* __restrict__ attn16) {
  int bb = blockIdx.z;
  _Float16* attnb = attn16 + (size_t)bb * S_LEN * S_LEN;
  int bm = blockIdx.y, bn = blockIdx.x;
  if (bn > bm) {  // entire 128x128 tile strictly above diagonal -> zeros
    _Float16* dst = attnb + (size_t)(bm * 128) * S_LEN + bn * 128;
    v8h z = {};
    for (int idx = threadIdx.x; idx < 128 * 16; idx += 256) {
      int rr = idx >> 4, cc = (idx & 15) * 8;
      *reinterpret_cast<v8h*>(dst + (size_t)rr * S_LEN + cc) = z;
    }
    return;
  }
  int tid = threadIdx.x, wave = tid >> 5, lane = tid & 31;
  int l16 = lane & 15, lh = lane >> 4;
  int m0 = bm * 128 + (wave & 3) * 32;
  int n0 = bn * 128 + (wave >> 2) * 64;
  const _Float16* Qb = q16 + (size_t)bb * S_LEN * ZDIM;
  const _Float16* Kb = k16 + (size_t)bb * S_LEN * ZDIM;
  v8f acc[2][4];
#pragma unroll
  for (int i = 0; i < 2; ++i)
#pragma unroll
    for (int j = 0; j < 4; ++j) acc[i][j] = (v8f){0,0,0,0,0,0,0,0};

  mma_loop<2, 4>(Qb, ZDIM, Kb, ZDIM, m0, n0, 0, ZDIM, l16, lh, acc);

  const float inv_s = 1.0f / (float)S_LEN;
#pragma unroll
  for (int i = 0; i < 2; ++i)
#pragma unroll
    for (int j = 0; j < 4; ++j) {
      int kk = n0 + 16 * j + l16;
#pragma unroll
      for (int r = 0; r < 8; ++r) {
        int qq = m0 + 16 * i + lh * 8 + r;
        float val = acc[i][j][r] * inv_s + rpb[(S_LEN - 1) + kk - qq];
        val = (kk > qq) ? 0.0f : fmaxf(val, 0.0f);
        attnb[(size_t)qq * S_LEN + kk] = (_Float16)(val * val);
      }
    }
}

// ---- v (B,S,H) -> vT (B,H,S) via LDS ---------------------------------------
__global__ __launch_bounds__(256) void transpose_v(const _Float16* __restrict__ v,
                                                   _Float16* __restrict__ vt) {
  __shared__ _Float16 tile[32][33];
  int bb = blockIdx.z, s0 = blockIdx.y * 32, h0 = blockIdx.x * 32;
  int x = threadIdx.x, y = threadIdx.y;
  const _Float16* src = v + ((size_t)bb * S_LEN + s0) * HDIM + h0;
#pragma unroll
  for (int j = 0; j < 4; ++j) tile[y + 8 * j][x] = src[(size_t)(y + 8 * j) * HDIM + x];
  __syncthreads();
  _Float16* dst = vt + ((size_t)bb * HDIM + h0) * S_LEN + s0;
#pragma unroll
  for (int j = 0; j < 4; ++j) dst[(size_t)(y + 8 * j) * S_LEN + x] = tile[x][y + 8 * j];
}

// ---- attn @ v, fused *u, f16 out (rows re-laid as s*BSZ+b) -----------------
__global__ __launch_bounds__(256) void av_kernel(const _Float16* __restrict__ attn16,
                                                 const _Float16* __restrict__ vT16,
                                                 const _Float16* __restrict__ u16,
                                                 _Float16* __restrict__ g16) {
  int bb = blockIdx.z;
  int tid = threadIdx.x, wave = tid >> 5, lane = tid & 31;
  int l16 = lane & 15, lh = lane >> 4;
  int m0 = blockIdx.y * 128 + (wave & 3) * 32;
  int n0 = blockIdx.x * 128 + (wave >> 2) * 64;
  const _Float16* Ab = attn16 + (size_t)bb * S_LEN * S_LEN;
  const _Float16* Bb = vT16 + (size_t)bb * HDIM * S_LEN;
  int kend = imin(S_LEN, (blockIdx.y + 1) * 128);  // causal: attn==0 beyond diag
  v8f acc[2][4];
#pragma unroll
  for (int i = 0; i < 2; ++i)
#pragma unroll
    for (int j = 0; j < 4; ++j) acc[i][j] = (v8f){0,0,0,0,0,0,0,0};

  mma_loop<2, 4>(Ab, S_LEN, Bb, S_LEN, m0, n0, 0, kend, l16, lh, acc);

#pragma unroll
  for (int i = 0; i < 2; ++i)
#pragma unroll
    for (int j = 0; j < 4; ++j) {
      int h = n0 + 16 * j + l16;
#pragma unroll
      for (int r = 0; r < 8; ++r) {
        int s = m0 + 16 * i + lh * 8 + r;
        size_t idx = ((size_t)s * BSZ + bb) * HDIM + h;
        g16[idx] = (_Float16)(acc[i][j][r] * (float)u16[idx]);
      }
    }
}

// ---- final: out = g @ out_w^T + out_b (f32) --------------------------------
__global__ __launch_bounds__(256) void out_kernel(const _Float16* __restrict__ g16,
                                                  const _Float16* __restrict__ owT,
                                                  const float* __restrict__ out_b,
                                                  float* __restrict__ out) {
  int tid = threadIdx.x, wave = tid >> 5, lane = tid & 31;
  int l16 = lane & 15, lh = lane >> 4;
  int m0 = blockIdx.y * 128 + (wave & 3) * 32;
  int n0 = blockIdx.x * 128 + (wave >> 2) * 64;
  v8f acc[2][4];
#pragma unroll
  for (int i = 0; i < 2; ++i)
#pragma unroll
    for (int j = 0; j < 4; ++j) acc[i][j] = (v8f){0,0,0,0,0,0,0,0};

  mma_loop<2, 4>(g16, HDIM, owT, HDIM, m0, n0, 0, HDIM, l16, lh, acc);

#pragma unroll
  for (int i = 0; i < 2; ++i)
#pragma unroll
    for (int j = 0; j < 4; ++j) {
      int n = n0 + 16 * j + l16;
#pragma unroll
      for (int r = 0; r < 8; ++r) {
        int m = m0 + 16 * i + lh * 8 + r;
        out[(size_t)m * EDIM + n] = acc[i][j][r] + out_b[n];
      }
    }
}

// ---- workspace layout (bytes) ---------------------------------------------
#define OFF_X16   ((size_t)0)
#define SZ_X16    ((size_t)M1 * EDIM * 2)
#define OFF_PW16  (OFF_X16 + SZ_X16)
#define SZ_PW16   ((size_t)N1 * EDIM * 2)
#define OFF_OW16  (OFF_PW16 + SZ_PW16)
#define SZ_OW16   ((size_t)EDIM * HDIM * 2)
#define OFF_U16   (OFF_OW16 + SZ_OW16)
#define SZ_U16    ((size_t)M1 * HDIM * 2)
#define OFF_V16   (OFF_U16 + SZ_U16)                 // also reused as g16
#define SZ_V16    ((size_t)BSZ * S_LEN * HDIM * 2)
#define OFF_VT16  (OFF_V16 + SZ_V16)
#define SZ_VT16   SZ_V16
#define OFF_Q16   (OFF_VT16 + SZ_VT16)
#define SZ_Q16    ((size_t)BSZ * S_LEN * ZDIM * 2)
#define OFF_K16   (OFF_Q16 + SZ_Q16)
#define OFF_ATTN  (OFF_K16 + SZ_Q16)

extern "C" void kernel_launch(void* const* d_in, const int* in_sizes, int n_in,
                              void* d_out, int out_size, void* d_ws, size_t ws_size,
                              hipStream_t stream) {
  const float* x      = (const float*)d_in[0];
  // d_in[1] = attn_mask (causal triu) is reimplemented analytically
  const float* proj_w = (const float*)d_in[2];
  const float* proj_b = (const float*)d_in[3];
  const float* out_w  = (const float*)d_in[4];
  const float* out_b  = (const float*)d_in[5];
  const float* gamma  = (const float*)d_in[6];
  const float* beta   = (const float*)d_in[7];
  const float* rpb    = (const float*)d_in[8];
  float* out = (float*)d_out;

  char* ws = (char*)d_ws;
  _Float16* x16  = (_Float16*)(ws + OFF_X16);
  _Float16* pw16 = (_Float16*)(ws + OFF_PW16);
  _Float16* ow16 = (_Float16*)(ws + OFF_OW16);
  _Float16* u16  = (_Float16*)(ws + OFF_U16);
  _Float16* v16  = (_Float16*)(ws + OFF_V16);
  _Float16* vT16 = (_Float16*)(ws + OFF_VT16);
  _Float16* q16  = (_Float16*)(ws + OFF_Q16);
  _Float16* k16  = (_Float16*)(ws + OFF_K16);
  _Float16* at16 = (_Float16*)(ws + OFF_ATTN);
  _Float16* g16  = v16;  // v16 dead after transpose; reuse for gated output

  f32_to_f16<<<2048, 256, 0, stream>>>(x, x16, (M1 * EDIM) / 4);
  f32_to_f16<<<2048, 256, 0, stream>>>(proj_w, pw16, (N1 * EDIM) / 4);
  f32_to_f16<<<2048, 256, 0, stream>>>(out_w, ow16, (EDIM * HDIM) / 4);

  dim3 g1(N1 / 128, M1 / 128);                       // 33 x 64
  gemm1_kernel<<<g1, 256, 0, stream>>>(x16, pw16, proj_b, gamma, beta,
                                       u16, v16, q16, k16);

  dim3 gqk(S_LEN / 128, S_LEN / 128, BSZ);           // 16 x 16 x 4
  qk_kernel<<<gqk, 256, 0, stream>>>(q16, k16, rpb, at16);

  dim3 gt(HDIM / 32, S_LEN / 32, BSZ);               // 64 x 64 x 4
  transpose_v<<<gt, dim3(32, 8), 0, stream>>>(v16, vT16);

  dim3 gav(HDIM / 128, S_LEN / 128, BSZ);            // 16 x 16 x 4
  av_kernel<<<gav, 256, 0, stream>>>(at16, vT16, u16, g16);

  dim3 go(EDIM / 128, M1 / 128);                     // 8 x 64
  out_kernel<<<go, 256, 0, stream>>>(g16, ow16, out_b, out);
}
